// GNN_21638045237382
// MI455X (gfx1250) — matrix operand order repaired and checked by reading, not compiled, not run on hardware
//
#include <hip/hip_runtime.h>
#include <hip/hip_bf16.h>

typedef __attribute__((ext_vector_type(16))) __bf16 v16bf;
typedef __attribute__((ext_vector_type(8)))  float  v8f;

#define N_NODES 100000
#define NP      100096      /* padded to multiple of 256 */
#define N_EDGES 1600000
#define EMB     300
#define K1      320         /* padded K for GEMM1 */
#define HID     600
#define HP      640         /* padded hidden / padded K for GEMM2 */
#define LAYERS  5
#define BN_EPS  1e-5f

union Frag { uint4 q[2]; v16bf v; };

// ---------------- input embedding: h = xe1[x_atom] + xe2[x_chir] ----------------
__global__ void k_h0(const int* __restrict__ xa, const int* __restrict__ xc,
                     const float* __restrict__ xe1, const float* __restrict__ xe2,
                     float* __restrict__ h) {
    int node = blockIdx.x;
    const float* r1 = xe1 + (long)xa[node] * EMB;
    const float* r2 = xe2 + (long)xc[node] * EMB;
    float* out = h + (long)node * EMB;
    for (int f = threadIdx.x; f < EMB; f += blockDim.x)
        out[f] = r1[f] + r2[f];
}

// ------------- agg init with self-loop term: agg = h + ee1[4] + ee2[0] ----------
__global__ void k_agginit(const float* __restrict__ h, const float* __restrict__ e1row,
                          const float* __restrict__ e2row, float* __restrict__ agg) {
    int node = blockIdx.x;
    const float* hr = h + (long)node * EMB;
    float* ar = agg + (long)node * EMB;
    for (int f = threadIdx.x; f < EMB; f += blockDim.x)
        ar[f] = hr[f] + e1row[f] + e2row[f];
}

// ------------- edge messages + scatter-add (atomics stay in 192MB L2) -----------
__global__ void k_edges(const float* __restrict__ h,
                        const int* __restrict__ src, const int* __restrict__ dst,
                        const int* __restrict__ et, const int* __restrict__ ed,
                        const float* __restrict__ e1base, const float* __restrict__ e2base,
                        float* __restrict__ agg) {
    for (int e = blockIdx.x; e < N_EDGES; e += gridDim.x) {
        int s = src[e], d = dst[e];
        const float* e1 = e1base + (long)et[e] * EMB;
        const float* e2 = e2base + (long)ed[e] * EMB;
        const float* hs = h + (long)s * EMB;
        float* ad = agg + (long)d * EMB;
        for (int f = threadIdx.x; f < EMB; f += blockDim.x)
            atomicAdd(ad + f, hs[f] + e1[f] + e2[f]);
    }
}

// ------------- fp32 agg -> zero-padded bf16 activations [NP][K1] ----------------
__global__ void k_conv_h(const float* __restrict__ agg, __bf16* __restrict__ abf) {
    int node = blockIdx.x;          // grid = NP
    int k = threadIdx.x;            // blockDim = K1 (320)
    float v = 0.f;
    if (node < N_NODES && k < EMB) v = agg[(long)node * EMB + k];
    abf[(long)node * K1 + k] = (__bf16)v;
}

// ------------- weight conversion to transposed, padded bf16 ---------------------
__global__ void k_conv_w1(const float* __restrict__ W1, __bf16* __restrict__ w1t) {
    long total = (long)LAYERS * HP * K1;
    for (long i = (long)blockIdx.x * blockDim.x + threadIdx.x; i < total;
         i += (long)gridDim.x * blockDim.x) {
        int  k = (int)(i % K1);
        long t = i / K1;
        int  n = (int)(t % HP);
        int  l = (int)(t / HP);
        float v = 0.f;
        if (n < HID && k < EMB) v = W1[((long)l * EMB + k) * HID + n];
        w1t[i] = (__bf16)v;
    }
}

__global__ void k_conv_w2(const float* __restrict__ W2, __bf16* __restrict__ w2t) {
    long total = (long)LAYERS * K1 * HP;     // [L][320 cols][640 k]
    for (long i = (long)blockIdx.x * blockDim.x + threadIdx.x; i < total;
         i += (long)gridDim.x * blockDim.x) {
        int  k = (int)(i % HP);
        long t = i / HP;
        int  n = (int)(t % K1);
        int  l = (int)(t / K1);
        float v = 0.f;
        if (n < EMB && k < HID) v = W2[((long)l * HID + k) * EMB + n];
        w2t[i] = (__bf16)v;
    }
}

// ------------- fragment loaders -------------------------------------------------
__device__ __forceinline__ Frag loadA(const __bf16* ap, int k, int half) {
    // 16-bit A layout: half 0 holds K {0..7,16..23}, half 1 holds {8..15,24..31}
    Frag a;
    a.q[0] = *(const uint4*)(ap + k + half * 8);
    a.q[1] = *(const uint4*)(ap + k + 16 + half * 8);
    return a;
}
__device__ __forceinline__ Frag loadB(const __bf16* bp, int k) {
    // B layout: lanes 0-15 K {k..k+7,k+16..k+23} (col=ml), lanes 16-31 K {k+8..k+15,k+24..k+31}
    Frag b;
    b.q[0] = *(const uint4*)(bp + k);
    b.q[1] = *(const uint4*)(bp + k + 16);
    return b;
}

// ------------- pipelined wave-level 32x64 bf16 WMMA tile ------------------------
// 2 A-fragments x 4 B-fragments = 8 WMMAs per K-step; next step's fragments are
// prefetched before the current WMMAs issue so waits can be partial.
__device__ __forceinline__ void mma32x64(const __bf16* __restrict__ A,
                                         const __bf16* __restrict__ BT,
                                         long rowBase, int colBase, int Kpad,
                                         v8f acc[2][4]) {
    int lane = threadIdx.x & 31;
    int half = lane >> 4;       // 0: lanes 0-15, 1: lanes 16-31
    int ml   = lane & 15;
    const __bf16* ap0 = A + (rowBase + ml) * (long)Kpad;
    const __bf16* ap1 = ap0 + 16L * Kpad;
    const __bf16* bp0 = BT + (long)(colBase + ml) * Kpad + half * 8;
    long bs = 16L * Kpad;

    Frag a0 = loadA(ap0, 0, half);
    Frag a1 = loadA(ap1, 0, half);
    Frag b0 = loadB(bp0 + 0 * bs, 0);
    Frag b1 = loadB(bp0 + 1 * bs, 0);
    Frag b2 = loadB(bp0 + 2 * bs, 0);
    Frag b3 = loadB(bp0 + 3 * bs, 0);

    for (int k = 32; k < Kpad; k += 32) {
        Frag na0 = loadA(ap0, k, half);
        Frag na1 = loadA(ap1, k, half);
        Frag nb0 = loadB(bp0 + 0 * bs, k);
        Frag nb1 = loadB(bp0 + 1 * bs, k);
        Frag nb2 = loadB(bp0 + 2 * bs, k);
        Frag nb3 = loadB(bp0 + 3 * bs, k);
        acc[0][0] = __builtin_amdgcn_wmma_f32_16x16x32_bf16(false, a0.v, false, b0.v, (short)0, acc[0][0], false, false);
        acc[1][0] = __builtin_amdgcn_wmma_f32_16x16x32_bf16(false, a1.v, false, b0.v, (short)0, acc[1][0], false, false);
        acc[0][1] = __builtin_amdgcn_wmma_f32_16x16x32_bf16(false, a0.v, false, b1.v, (short)0, acc[0][1], false, false);
        acc[1][1] = __builtin_amdgcn_wmma_f32_16x16x32_bf16(false, a1.v, false, b1.v, (short)0, acc[1][1], false, false);
        acc[0][2] = __builtin_amdgcn_wmma_f32_16x16x32_bf16(false, a0.v, false, b2.v, (short)0, acc[0][2], false, false);
        acc[1][2] = __builtin_amdgcn_wmma_f32_16x16x32_bf16(false, a1.v, false, b2.v, (short)0, acc[1][2], false, false);
        acc[0][3] = __builtin_amdgcn_wmma_f32_16x16x32_bf16(false, a0.v, false, b3.v, (short)0, acc[0][3], false, false);
        acc[1][3] = __builtin_amdgcn_wmma_f32_16x16x32_bf16(false, a1.v, false, b3.v, (short)0, acc[1][3], false, false);
        a0 = na0; a1 = na1; b0 = nb0; b1 = nb1; b2 = nb2; b3 = nb3;
    }
    // peeled final K-step
    acc[0][0] = __builtin_amdgcn_wmma_f32_16x16x32_bf16(false, a0.v, false, b0.v, (short)0, acc[0][0], false, false);
    acc[1][0] = __builtin_amdgcn_wmma_f32_16x16x32_bf16(false, a1.v, false, b0.v, (short)0, acc[1][0], false, false);
    acc[0][1] = __builtin_amdgcn_wmma_f32_16x16x32_bf16(false, a0.v, false, b1.v, (short)0, acc[0][1], false, false);
    acc[1][1] = __builtin_amdgcn_wmma_f32_16x16x32_bf16(false, a1.v, false, b1.v, (short)0, acc[1][1], false, false);
    acc[0][2] = __builtin_amdgcn_wmma_f32_16x16x32_bf16(false, a0.v, false, b2.v, (short)0, acc[0][2], false, false);
    acc[1][2] = __builtin_amdgcn_wmma_f32_16x16x32_bf16(false, a1.v, false, b2.v, (short)0, acc[1][2], false, false);
    acc[0][3] = __builtin_amdgcn_wmma_f32_16x16x32_bf16(false, a0.v, false, b3.v, (short)0, acc[0][3], false, false);
    acc[1][3] = __builtin_amdgcn_wmma_f32_16x16x32_bf16(false, a1.v, false, b3.v, (short)0, acc[1][3], false, false);
}

// ------------- GEMM1: agg_bf16 x W1^T + b1, ReLU, -> bf16 hid [NP][HP] ----------
__global__ void __launch_bounds__(256) k_gemm1(const __bf16* __restrict__ abf,
                                               const __bf16* __restrict__ w1t,
                                               const float* __restrict__ b1l,
                                               __bf16* __restrict__ hidbf) {
    int wave = threadIdx.x >> 5;
    long rowBase = (long)blockIdx.x * 256 + wave * 32;
    int  colBase = blockIdx.y * 64;
    v8f z = {0.f, 0.f, 0.f, 0.f, 0.f, 0.f, 0.f, 0.f};
    v8f acc[2][4] = {{z, z, z, z}, {z, z, z, z}};
    mma32x64(abf, w1t, rowBase, colBase, K1, acc);
    int lane = threadIdx.x & 31, half = lane >> 4, ml = lane & 15;
#pragma unroll
    for (int i = 0; i < 2; ++i) {
#pragma unroll
        for (int t = 0; t < 4; ++t) {
            int col = colBase + t * 16 + ml;
            float bias = (col < HID) ? b1l[col] : 0.f;
#pragma unroll
            for (int r = 0; r < 8; ++r) {
                long row = rowBase + i * 16 + r + 8 * half;
                float v = fmaxf(acc[i][t][r] + bias, 0.f);
                if (col >= HID) v = 0.f;            // keep padding columns zero
                hidbf[row * HP + col] = (__bf16)v;
            }
        }
    }
}

// ------------- GEMM2: hid_bf16 x W2^T + b2 -> fp32 h [N][EMB] -------------------
__global__ void __launch_bounds__(256) k_gemm2(const __bf16* __restrict__ hidbf,
                                               const __bf16* __restrict__ w2t,
                                               const float* __restrict__ b2l,
                                               float* __restrict__ hout) {
    int wave = threadIdx.x >> 5;
    long rowBase = (long)blockIdx.x * 256 + wave * 32;
    int  colBase = blockIdx.y * 64;
    v8f z = {0.f, 0.f, 0.f, 0.f, 0.f, 0.f, 0.f, 0.f};
    v8f acc[2][4] = {{z, z, z, z}, {z, z, z, z}};
    mma32x64(hidbf, w2t, rowBase, colBase, HP, acc);
    int lane = threadIdx.x & 31, half = lane >> 4, ml = lane & 15;
#pragma unroll
    for (int i = 0; i < 2; ++i) {
#pragma unroll
        for (int t = 0; t < 4; ++t) {
            int col = colBase + t * 16 + ml;
            if (col >= EMB) continue;
            float bias = b2l[col];
#pragma unroll
            for (int r = 0; r < 8; ++r) {
                long row = rowBase + i * 16 + r + 8 * half;
                if (row < N_NODES)
                    hout[row * EMB + col] = acc[i][t][r] + bias;
            }
        }
    }
}

// ------------- BatchNorm statistics ---------------------------------------------
__global__ void k_zero(float* __restrict__ p, int n) {
    int i = blockIdx.x * blockDim.x + threadIdx.x;
    if (i < n) p[i] = 0.f;
}

__global__ void k_bnred(const float* __restrict__ h, float* __restrict__ sums) {
    int t = threadIdx.x;                 // 256 threads: features t and 256+t
    int f2 = 256 + t;
    bool has2 = f2 < EMB;
    float s1 = 0.f, q1 = 0.f, s2 = 0.f, q2 = 0.f;
    for (int row = blockIdx.x; row < N_NODES; row += gridDim.x) {
        const float* hr = h + (long)row * EMB;
        float v = hr[t];  s1 += v;  q1 += v * v;
        if (has2) { float w = hr[f2]; s2 += w; q2 += w * w; }
    }
    atomicAdd(&sums[t], s1);
    atomicAdd(&sums[EMB + t], q1);
    if (has2) {
        atomicAdd(&sums[f2], s2);
        atomicAdd(&sums[EMB + f2], q2);
    }
}

__global__ void k_bnapply(float* __restrict__ h, const float* __restrict__ sums,
                          const float* __restrict__ gl, const float* __restrict__ bl,
                          int relu) {
    int node = blockIdx.x;
    const float invN = 1.f / (float)N_NODES;
    for (int f = threadIdx.x; f < EMB; f += blockDim.x) {
        float mean = sums[f] * invN;
        float var  = sums[EMB + f] * invN - mean * mean;
        float inv  = rsqrtf(var + BN_EPS);
        float v = (h[(long)node * EMB + f] - mean) * inv * gl[f] + bl[f];
        if (relu) v = fmaxf(v, 0.f);
        h[(long)node * EMB + f] = v;
    }
}

// --------------------------------------------------------------------------------
extern "C" void kernel_launch(void* const* d_in, const int* in_sizes, int n_in,
                              void* d_out, int out_size, void* d_ws, size_t ws_size,
                              hipStream_t stream) {
    const int*   x_atom = (const int*)d_in[0];
    const int*   x_chir = (const int*)d_in[1];
    const int*   eidx   = (const int*)d_in[2];
    const int*   ea_t   = (const int*)d_in[3];
    const int*   ea_d   = (const int*)d_in[4];
    const float* xe1    = (const float*)d_in[5];
    const float* xe2    = (const float*)d_in[6];
    const float* ee1    = (const float*)d_in[7];
    const float* ee2    = (const float*)d_in[8];
    const float* W1     = (const float*)d_in[9];
    const float* b1     = (const float*)d_in[10];
    const float* W2     = (const float*)d_in[11];
    const float* b2     = (const float*)d_in[12];
    const float* gamma  = (const float*)d_in[13];
    const float* beta   = (const float*)d_in[14];

    float* h = (float*)d_out;            // h lives in d_out (fully rewritten)
    const int* src = eidx;
    const int* dst = eidx + N_EDGES;

    // workspace carving (256B aligned)
    char* p = (char*)d_ws;
    auto carve = [&](size_t bytes) -> void* {
        void* r = (void*)p;
        p += (bytes + 255) & ~(size_t)255;
        return r;
    };
    float*  agg   = (float*) carve((size_t)N_NODES * EMB * sizeof(float));
    __bf16* abf   = (__bf16*)carve((size_t)NP * K1 * sizeof(__bf16));
    __bf16* hidbf = (__bf16*)carve((size_t)NP * HP * sizeof(__bf16));
    __bf16* w1t   = (__bf16*)carve((size_t)LAYERS * HP * K1 * sizeof(__bf16));
    __bf16* w2t   = (__bf16*)carve((size_t)LAYERS * K1 * HP * sizeof(__bf16));
    float*  sums  = (float*) carve(2 * EMB * sizeof(float));

    k_conv_w1<<<512, 256, 0, stream>>>(W1, w1t);
    k_conv_w2<<<512, 256, 0, stream>>>(W2, w2t);
    k_h0<<<N_NODES, 256, 0, stream>>>(x_atom, x_chir, xe1, xe2, h);

    for (int l = 0; l < LAYERS; ++l) {
        const float* e1b = ee1 + (size_t)l * 6 * EMB;
        const float* e2b = ee2 + (size_t)l * 3 * EMB;
        k_agginit<<<N_NODES, 256, 0, stream>>>(h, e1b + 4 * EMB, e2b, agg);
        k_edges<<<16384, 256, 0, stream>>>(h, src, dst, ea_t, ea_d, e1b, e2b, agg);
        k_conv_h<<<NP, K1, 0, stream>>>(agg, abf);
        k_gemm1<<<dim3(NP / 256, HP / 64), 256, 0, stream>>>(
            abf, w1t + (size_t)l * HP * K1, b1 + (size_t)l * HID, hidbf);
        k_gemm2<<<dim3(NP / 256, K1 / 64), 256, 0, stream>>>(
            hidbf, w2t + (size_t)l * K1 * HP, b2 + (size_t)l * EMB, h);
        k_zero<<<3, 256, 0, stream>>>(sums, 2 * EMB);
        k_bnred<<<512, 256, 0, stream>>>(h, sums);
        k_bnapply<<<N_NODES, 256, 0, stream>>>(
            h, sums, gamma + (size_t)l * EMB, beta + (size_t)l * EMB,
            (l < LAYERS - 1) ? 1 : 0);
    }
}
